// MambaGPT_77412490543260
// MI455X (gfx1250) — matrix operand order repaired and checked by reading, not compiled, not run on hardware
//
#include <hip/hip_runtime.h>
#include <hip/hip_bf16.h>

// ---------------- model constants (match reference) ----------------
#define D_MODEL   512
#define N_LAYERS  12
#define D_STATE   64
#define D_CONV    4
#define HEADDIM   64
#define SEQ       2048
#define B_SZ      2
#define D_INNER   1024            // EXPAND * D_MODEL
#define NHEADS    16              // D_INNER / HEADDIM
#define CONV_DIM  1152            // D_INNER + 2*D_STATE
#define D_IN_PROJ 2192            // 2*D_INNER + 2*D_STATE + NHEADS
#define CHUNK     128
#define NCHUNK    (SEQ / CHUNK)   // 16
#define KLEN3     30
#define EPS       1e-5f
#define ROWS      (B_SZ * SEQ)    // 4096

typedef __attribute__((ext_vector_type(16))) _Float16 v16h;
typedef __attribute__((ext_vector_type(8)))  _Float16 v8h;
typedef __attribute__((ext_vector_type(8)))  float    v8f;

// v_wmma_f32_16x16x32_f16 : D(16x16 f32) = A(16x32 f16) * B(32x16 f16) + C
__device__ __forceinline__ v8f wmma_f16(v16h a, v16h b, v8f c) {
  return __builtin_amdgcn_wmma_f32_16x16x32_f16(false, a, false, b, (short)0, c,
                                                false, false);
}
__device__ __forceinline__ float sigmoid_f(float x) { return 1.f / (1.f + __expf(-x)); }

__device__ __forceinline__ v16h combine16(v8h lo, v8h hi) {
  return __builtin_shufflevector(lo, hi, 0, 1, 2, 3, 4, 5, 6, 7,
                                 8, 9, 10, 11, 12, 13, 14, 15);
}
__device__ __forceinline__ v16h pack16(float4 q0, float4 q1, float4 q2, float4 q3) {
  v16h a;
  a[0]  = (_Float16)q0.x; a[1]  = (_Float16)q0.y; a[2]  = (_Float16)q0.z; a[3]  = (_Float16)q0.w;
  a[4]  = (_Float16)q1.x; a[5]  = (_Float16)q1.y; a[6]  = (_Float16)q1.z; a[7]  = (_Float16)q1.w;
  a[8]  = (_Float16)q2.x; a[9]  = (_Float16)q2.y; a[10] = (_Float16)q2.z; a[11] = (_Float16)q2.w;
  a[12] = (_Float16)q3.x; a[13] = (_Float16)q3.y; a[14] = (_Float16)q3.z; a[15] = (_Float16)q3.w;
  return a;
}
// A-fragment from fp32 row: runs at p[0..7] and p[16..23]  (lane&15 = M row)
__device__ __forceinline__ v16h load_afrag_f32(const float* p) {
  return pack16(*(const float4*)p, *(const float4*)(p + 4),
                *(const float4*)(p + 16), *(const float4*)(p + 20));
}
// B-fragment from fp32 row: 16 contiguous values (lane&15 = N col; K = ((lane>>4)<<4)+e)
__device__ __forceinline__ v16h load_bfrag_f32(const float* p) {
  return pack16(*(const float4*)p, *(const float4*)(p + 4),
                *(const float4*)(p + 8), *(const float4*)(p + 12));
}
__device__ __forceinline__ v8h lds8(const _Float16* p) { return *(const v8h*)p; }

// ---------------- embed + NeRF positional encoding ----------------
__global__ void embed_kernel(const float* __restrict__ x, const float* __restrict__ Wemb,
                             float* __restrict__ h) {
  int idx = blockIdx.x * blockDim.x + threadIdx.x;
  if (idx >= ROWS * D_MODEL) return;
  int d = idx & (D_MODEL - 1);
  int row = idx >> 9;
  float xm[3];
#pragma unroll
  for (int k = 0; k < 3; ++k) {
    float v = x[row * 3 + k];
    xm[k] = (v == -100.0f) ? 0.0f : v;
  }
  float acc = 0.f;
#pragma unroll
  for (int k = 0; k < 3; ++k) acc += xm[k] * Wemb[k * D_MODEL + d];
  if (d < 510) {                       // nf = 512/(2*3) = 85; pe dim = 3*170 = 510
    int k = d / 170, t = d % 170;
    float fr = exp2f((float)(t < 85 ? t : t - 85));
    float ang = xm[k] * fr;
    acc += (t < 85) ? sinf(ang) : cosf(ang);
  }
  h[idx] = acc;
}

// ---------------- rmsnorm (one block per row) ----------------
__global__ void rmsnorm_kernel(const float* __restrict__ x, const float* __restrict__ w,
                               float* __restrict__ out, int D) {
  int row = blockIdx.x, tid = threadIdx.x;
  __shared__ float red[256];
  const float* xr = x + (size_t)row * D;
  float ss = 0.f;
  for (int i = tid; i < D; i += 256) { float v = xr[i]; ss += v * v; }
  red[tid] = ss; __syncthreads();
  for (int s = 128; s > 0; s >>= 1) { if (tid < s) red[tid] += red[tid + s]; __syncthreads(); }
  float inv = rsqrtf(red[0] / (float)D + EPS);
  for (int i = tid; i < D; i += 256) out[(size_t)row * D + i] = xr[i] * inv * w[i];
}

// ---------------- WMMA GEMM: C[M,N] (+)= A[M,K] * B[K,N], fp32 in, f16 WMMA ----------
// block = 128 threads = 4 waves; block tile 64(M) x 64(N).
// Each thread stages one full 16-element B fragment row: 16 column loads
// (coalesced across threads) -> v_cvt_pk -> two ds_store_b128. Consumers read
// fragments with two ds_load_b128. Double-buffered: one barrier per K-step.
// N-edge handled branchlessly (clamped pointer + 0/1 scale) -> no EXEC juggling.
template <int ADDC>
__global__ __launch_bounds__(128) void gemm_wmma_kernel(
    const float* __restrict__ A, const float* __restrict__ B, float* __restrict__ C,
    int N, int K, int lda, int ldb, int ldc)
{
  __shared__ __align__(32) _Float16 BsF[2][4][32][16];  // [buf][ntile][lane][e] = 8 KB
  const int tid = threadIdx.x;
  const int wv = tid >> 5, lane = tid & 31, ln = lane & 15;
  const int n0 = blockIdx.x * 64, m0 = blockIdx.y * 64;
  const int kgA = (lane >> 4) << 3;   // 0 or 8
  // staging role: thread owns column sn and k-subgroup skg of the 32x64 B tile
  const int sn    = tid & 63;
  const int skg   = (tid >> 6) << 4;        // 0 or 16
  const int stile = sn >> 4;
  const int slane = skg | (sn & 15);
  const int gcol  = n0 + sn;
  const float colscale = (gcol < N) ? 1.f : 0.f;
  const float* Bcol = B + (gcol < N ? gcol : 0);

  v8f acc[4];
#pragma unroll
  for (int t = 0; t < 4; ++t) { v8f z = {}; acc[t] = z; }
  const float* Arow = A + (size_t)(m0 + wv * 16 + ln) * lda;

  const int nk = K >> 5;
  for (int kk = 0; kk < nk; ++kk) {
    const int k0 = kk << 5;
    const int buf = kk & 1;
    float f[16];
#pragma unroll
    for (int j = 0; j < 16; ++j)
      f[j] = Bcol[(size_t)(k0 + skg + j) * ldb] * colscale;
    v8h lo, hi;
#pragma unroll
    for (int j = 0; j < 8; ++j) { lo[j] = (_Float16)f[j]; hi[j] = (_Float16)f[j + 8]; }
    v8h* dst = (v8h*)&BsF[buf][stile][slane][0];
    dst[0] = lo; dst[1] = hi;
    if (kk + 1 < nk)   // stream next tile toward L2 (global_prefetch_b8)
      __builtin_prefetch(Bcol + (size_t)(k0 + 32 + skg) * ldb, 0, 1);
    __syncthreads();

    v16h a = load_bfrag_f32(Arow + k0 + ((lane >> 4) << 4));
#pragma unroll
    for (int t = 0; t < 4; ++t) {
      const _Float16* bp = &BsF[buf][t][lane][0];
      v16h bb = combine16(lds8(bp), lds8(bp + 8));
      acc[t] = wmma_f16(a, bb, acc[t]);
    }
  }
#pragma unroll
  for (int t = 0; t < 4; ++t) {
    int col = n0 + t * 16 + ln;
    if (col < N) {
#pragma unroll
      for (int r = 0; r < 8; ++r) {
        int row = m0 + wv * 16 + r + kgA;
        size_t o = (size_t)row * ldc + col;
        float v = acc[t][r];
        if (ADDC) C[o] += v; else C[o] = v;
      }
    }
  }
}

// ---------------- depthwise causal conv4 + SiLU + softplus(dt) ----------------
__global__ void conv_silu_dt_kernel(const float* __restrict__ zx,
                                    const float* __restrict__ cw,
                                    const float* __restrict__ cb,
                                    const float* __restrict__ dtbias,
                                    float* __restrict__ xcout, float* __restrict__ dtout)
{
  int row = blockIdx.x;               // b*SEQ + l
  int l = row & (SEQ - 1);
  int tid = threadIdx.x;
  for (int ch = tid; ch < CONV_DIM; ch += 256) {
    float acc = cb[ch];
#pragma unroll
    for (int k = 0; k < D_CONV; ++k) {
      int ll = l - (D_CONV - 1) + k;
      if (ll >= 0)
        acc += cw[ch * D_CONV + k] * zx[(size_t)(row - l + ll) * D_IN_PROJ + D_INNER + ch];
    }
    acc = acc * sigmoid_f(acc);
    xcout[(size_t)row * CONV_DIM + ch] = acc;
  }
  if (tid < NHEADS) {
    float v = zx[(size_t)row * D_IN_PROJ + D_INNER + CONV_DIM + tid] + dtbias[tid];
    dtout[(size_t)row * NHEADS + tid] = (v > 20.f) ? v : log1pf(__expf(v));
  }
}

// ---------------- fused SSD chunk kernel: one block per (b,c,h) ----------------
// Phase 0: cumsum(dt*A); Phase 1: M = (Cm Bm^T) * L * dt  (WMMA -> LDS f16)
// Phase 2: y_diag = M @ xh (+ D*xh)     Phase 3: S = (w . xh)^T @ Bm
// LDS: M-matrix (aliased to Bm^T for phase 3) + xh^T + small arrays, ~54.5 KB.
__global__ __launch_bounds__(128) void ssd_chunk_kernel(
    const float* __restrict__ xc, const float* __restrict__ dtb,
    const float* __restrict__ A_log, const float* __restrict__ Dp,
    float* __restrict__ y, float* __restrict__ S,
    float* __restrict__ dec, float* __restrict__ acum)
{
  int h = blockIdx.x, c = blockIdx.y, b = blockIdx.z;
  int tid = threadIdx.x, wv = tid >> 5, lane = tid & 31, ln = lane & 15;
  const int kgA = (lane >> 4) << 3;
  const int kgB = (lane >> 4) << 4;
  __shared__ __align__(16) _Float16 sMraw[128 * 136];  // M (and later Bm^T) 34.0 KB
  __shared__ __align__(16) _Float16 sXT[64][144];      // xh^T              18.0 KB
  __shared__ __align__(16) _Float16 sWh[128];
  __shared__ __align__(16) float sAc[128];
  __shared__ __align__(16) float sDt[128];
#define SM(i, j)  sMraw[(i) * 136 + (j)]
#define SBT(n, j) sMraw[(n) * 144 + (j)]
  const int rowbase = b * SEQ + c * CHUNK;
  const float Ah  = -__expf(A_log[h]);
  const float Dph = Dp[h];

  // ---- phase 0: cumsum of dt*A (Hillis-Steele) + xh^T staging ----
  sDt[tid] = dtb[(size_t)(rowbase + tid) * NHEADS + h];
  sAc[tid] = sDt[tid] * Ah;
  __syncthreads();
#pragma unroll
  for (int off = 1; off < 128; off <<= 1) {
    float v = (tid >= off) ? sAc[tid - off] : 0.f;
    __syncthreads();
    sAc[tid] += v;
    __syncthreads();
  }
  float asum = sAc[127];
  sWh[tid] = (_Float16)(__expf(asum - sAc[tid]) * sDt[tid]);
  acum[((size_t)(b * NCHUNK + c) * NHEADS + h) * CHUNK + tid] = sAc[tid];
  if (tid == 0) dec[(size_t)(b * NCHUNK + c) * NHEADS + h] = __expf(asum);
  {
    const float* xrow = xc + (size_t)(rowbase + tid) * CONV_DIM + h * HEADDIM;
    for (int p = 0; p < HEADDIM; ++p) sXT[p][tid] = (_Float16)xrow[p];  // transposed
  }
  __syncthreads();

  const float* Bg = xc + (size_t)rowbase * CONV_DIM + D_INNER;   // Bm[j][n]
  const float* Cg = Bg + D_STATE;                                // Cm[i][n]

  // ---- phase 1: scores -> masked/scaled M in LDS (branchless epilogue) ----
  for (int ti = wv; ti < 8; ti += 4) {
    int i0 = ti * 16;
    for (int tj = 0; tj < 8; ++tj) {
      int j0 = tj * 16;
      if (tj > ti) {   // strictly above diagonal: zero
#pragma unroll
        for (int r = 0; r < 8; ++r) SM(i0 + r + kgA, j0 + ln) = (_Float16)0.f;
        continue;
      }
      v8f acc = {};
#pragma unroll
      for (int k0 = 0; k0 < D_STATE; k0 += 32) {
        v16h a  = load_bfrag_f32(Cg + (size_t)(i0 + ln) * CONV_DIM + k0 + kgB);
        v16h bb = load_bfrag_f32(Bg + (size_t)(j0 + ln) * CONV_DIM + k0 + kgB);
        acc = wmma_f16(a, bb, acc);
      }
      // hoisted per-tile LDS reads (vectorized); causal mask only on diagonal
      // tiles, applied as a cndmask on the exp *argument* (exp(-100) -> 0).
      const float dtj = sDt[j0 + ln];
      const float acj = sAc[j0 + ln];
      const float4 ai0 = *(const float4*)&sAc[i0 + kgA];
      const float4 ai1 = *(const float4*)&sAc[i0 + kgA + 4];
      const float ai[8] = {ai0.x, ai0.y, ai0.z, ai0.w, ai1.x, ai1.y, ai1.z, ai1.w};
      const bool diag = (ti == tj);
      const int rth = ln - kgA;        // on diag tile: i>=j  <=>  r>=rth
#pragma unroll
      for (int r = 0; r < 8; ++r) {
        float d = ai[r] - acj;
        if (diag) d = (r >= rth) ? d : -100.f;
        float v = acc[r] * __expf(d) * dtj;
        SM(i0 + r + kgA, j0 + ln) = (_Float16)v;
      }
    }
  }
  __syncthreads();

  // ---- phase 2: y_diag = M @ xh  (+ D*xh) ----
  for (int ti = wv; ti < 8; ti += 4) {
    int i0 = ti * 16;
#pragma unroll
    for (int tp = 0; tp < 4; ++tp) {
      v8f acc = {};
#pragma unroll
      for (int k0 = 0; k0 < CHUNK; k0 += 32) {
        const _Float16* ap = &SM(i0 + ln, k0 + kgB);
        v16h a = combine16(lds8(ap), lds8(ap + 8));
        const _Float16* bp = &sXT[tp * 16 + ln][k0 + kgB];
        v16h bb = combine16(lds8(bp), lds8(bp + 8));
        acc = wmma_f16(a, bb, acc);
      }
      // D-term: xh[i][p] for i = i0+kgA..+7 is one contiguous v8h in sXT[p][]
      v8h xv8 = lds8(&sXT[tp * 16 + ln][i0 + kgA]);
#pragma unroll
      for (int r = 0; r < 8; ++r) {
        int i = i0 + r + kgA, p = tp * 16 + ln;
        y[(size_t)(rowbase + i) * D_INNER + h * HEADDIM + p] =
            acc[r] + Dph * (float)xv8[r];
      }
    }
  }
  __syncthreads();   // M is dead; reuse its LDS for Bm^T

  {
    const float* brow = Bg + (size_t)tid * CONV_DIM;
    for (int n = 0; n < D_STATE; ++n) SBT(n, tid) = (_Float16)brow[n];  // Bm^T
  }
  __syncthreads();

  // ---- phase 3: S[p,n] = sum_j w[j]*xh[j][p]*Bm[j][n] ----
  {
    int tp = wv;
    int pcol = tp * 16 + ln;
#pragma unroll
    for (int tn = 0; tn < 4; ++tn) {
      v8f acc = {};
#pragma unroll
      for (int j0 = 0; j0 < CHUNK; j0 += 32) {
        v8h xlo = lds8(&sXT[pcol][j0 + kgA]);
        v8h xhi = lds8(&sXT[pcol][j0 + kgA + 16]);
        v8h wlo = lds8(&sWh[j0 + kgA]);
        v8h whi = lds8(&sWh[j0 + kgA + 16]);
        v16h a = combine16(xlo * wlo, xhi * whi);       // v_pk_mul_f16
        const _Float16* bp = &SBT(tn * 16 + ln, j0 + kgA);
        v16h bb = combine16(lds8(bp), lds8(bp + 16));
        acc = wmma_f16(a, bb, acc);
      }
      size_t sbase = ((size_t)(b * NCHUNK + c) * NHEADS + h) * (HEADDIM * D_STATE);
#pragma unroll
      for (int r = 0; r < 8; ++r) {
        int p = tp * 16 + r + kgA, n = tn * 16 + ln;
        S[sbase + (size_t)p * D_STATE + n] = acc[r];
      }
    }
  }
#undef SM
#undef SBT
}

// ---------------- inter-chunk sequential scan: hs[c] = state before chunk c ------
__global__ void scan_kernel(const float* __restrict__ S, const float* __restrict__ dec,
                            float* __restrict__ hs)
{
  int h = blockIdx.x, b = blockIdx.y, tid = threadIdx.x;  // 256 threads, 16 states each
  float st[16];
#pragma unroll
  for (int i = 0; i < 16; ++i) st[i] = 0.f;
  for (int c = 0; c < NCHUNK; ++c) {
    size_t base = ((size_t)(b * NCHUNK + c) * NHEADS + h) * (HEADDIM * D_STATE) + (size_t)tid * 16;
    float d = dec[(size_t)(b * NCHUNK + c) * NHEADS + h];
#pragma unroll
    for (int i = 0; i < 16; ++i) {
      float s = S[base + i];
      hs[base + i] = st[i];
      st[i] = st[i] * d + s;
    }
  }
}

// ---------------- y += exp(a_cum) * (Cm @ hs^T) ----------------
__global__ __launch_bounds__(128) void yoff_kernel(
    const float* __restrict__ xc, const float* __restrict__ hs,
    const float* __restrict__ acum, float* __restrict__ y)
{
  int h = blockIdx.x, c = blockIdx.y, b = blockIdx.z;
  int tid = threadIdx.x, wv = tid >> 5, lane = tid & 31, ln = lane & 15;
  const int kgA = (lane >> 4) << 3, kgB = (lane >> 4) << 4;
  __shared__ __align__(16) float sE[128];
  sE[tid] = __expf(acum[((size_t)(b * NCHUNK + c) * NHEADS + h) * CHUNK + tid]);
  __syncthreads();
  const int rowbase = b * SEQ + c * CHUNK;
  const float* Cg = xc + (size_t)rowbase * CONV_DIM + D_INNER + D_STATE;
  const float* hb = hs + ((size_t)(b * NCHUNK + c) * NHEADS + h) * (HEADDIM * D_STATE);
  for (int ti = wv; ti < 8; ti += 4) {
    int i0 = ti * 16;
#pragma unroll
    for (int tp = 0; tp < 4; ++tp) {
      v8f acc = {};
#pragma unroll
      for (int k0 = 0; k0 < D_STATE; k0 += 32) {
        v16h a  = load_bfrag_f32(Cg + (size_t)(i0 + ln) * CONV_DIM + k0 + kgB);
        v16h bb = load_bfrag_f32(hb + (size_t)(tp * 16 + ln) * D_STATE + k0 + kgB);
        acc = wmma_f16(a, bb, acc);
      }
      const float4 e0 = *(const float4*)&sE[i0 + kgA];
      const float4 e1 = *(const float4*)&sE[i0 + kgA + 4];
      const float ei[8] = {e0.x, e0.y, e0.z, e0.w, e1.x, e1.y, e1.z, e1.w};
#pragma unroll
      for (int r = 0; r < 8; ++r) {
        int i = i0 + r + kgA, p = tp * 16 + ln;
        size_t o = (size_t)(rowbase + i) * D_INNER + h * HEADDIM + p;
        y[o] += ei[r] * acc[r];
      }
    }
  }
}

// ---------------- gated RMS norm: y = rmsnorm(y * silu(z)) * gw ----------------
__global__ void gate_norm_kernel(const float* __restrict__ zx,
                                 const float* __restrict__ gw,
                                 float* __restrict__ y)
{
  int row = blockIdx.x, tid = threadIdx.x;   // 256 threads, D_INNER=1024
  __shared__ float red[256];
  float vals[4];
  float ss = 0.f;
#pragma unroll
  for (int k = 0; k < 4; ++k) {
    int i = tid + k * 256;
    float z = zx[(size_t)row * D_IN_PROJ + i];
    float v = y[(size_t)row * D_INNER + i] * (z * sigmoid_f(z));
    vals[k] = v; ss += v * v;
  }
  red[tid] = ss; __syncthreads();
  for (int s = 128; s > 0; s >>= 1) { if (tid < s) red[tid] += red[tid + s]; __syncthreads(); }
  float inv = rsqrtf(red[0] / (float)D_INNER + EPS);
#pragma unroll
  for (int k = 0; k < 4; ++k) {
    int i = tid + k * 256;
    y[(size_t)row * D_INNER + i] = vals[k] * inv * gw[i];
  }
}

// ---------------- tiny output head (N=30, not WMMA-worthy) ----------------
__global__ void head_kernel(const float* __restrict__ xn, const float* __restrict__ hw,
                            const float* __restrict__ hb, float* __restrict__ out)
{
  int row = blockIdx.x, j = threadIdx.x;   // 32 threads
  if (j >= KLEN3) return;
  float acc = hb[j];
  const float* xr = xn + (size_t)row * D_MODEL;
  for (int k = 0; k < D_MODEL; ++k) acc += xr[k] * hw[k * KLEN3 + j];
  out[(size_t)row * KLEN3 + j] = acc;
}

// ---------------- host-side orchestration ----------------
extern "C" void kernel_launch(void* const* d_in, const int* in_sizes, int n_in,
                              void* d_out, int out_size, void* d_ws, size_t ws_size,
                              hipStream_t stream)
{
  (void)in_sizes; (void)n_in; (void)out_size; (void)ws_size;
  const float* x       = (const float*)d_in[0];
  const float* Wemb    = (const float*)d_in[1];
  const float* rms_w   = (const float*)d_in[2];
  const float* in_w    = (const float*)d_in[3];
  const float* conv_w  = (const float*)d_in[4];
  const float* conv_b  = (const float*)d_in[5];
  const float* dt_bias = (const float*)d_in[6];
  const float* A_log   = (const float*)d_in[7];
  const float* D_param = (const float*)d_in[8];
  const float* gnorm_w = (const float*)d_in[9];
  const float* out_w   = (const float*)d_in[10];
  const float* fnorm_w = (const float*)d_in[11];
  const float* head_w  = (const float*)d_in[12];
  const float* head_b  = (const float*)d_in[13];
  float* out = (float*)d_out;

  float* ws = (float*)d_ws;
  float* h     = ws; ws += (size_t)ROWS * D_MODEL;
  float* xn    = ws; ws += (size_t)ROWS * D_MODEL;
  float* zx    = ws; ws += (size_t)ROWS * D_IN_PROJ;
  float* xcb   = ws; ws += (size_t)ROWS * CONV_DIM;
  float* dtb   = ws; ws += (size_t)ROWS * NHEADS;
  float* ybuf  = ws; ws += (size_t)ROWS * D_INNER;
  float* Sbuf  = ws; ws += (size_t)B_SZ * NCHUNK * NHEADS * HEADDIM * D_STATE;
  float* hsbuf = ws; ws += (size_t)B_SZ * NCHUNK * NHEADS * HEADDIM * D_STATE;
  float* acum  = ws; ws += (size_t)B_SZ * NCHUNK * NHEADS * CHUNK;
  float* decb  = ws; ws += (size_t)B_SZ * NCHUNK * NHEADS;

  embed_kernel<<<(ROWS * D_MODEL + 255) / 256, 256, 0, stream>>>(x, Wemb, h);

  for (int l = 0; l < N_LAYERS; ++l) {
    rmsnorm_kernel<<<ROWS, 256, 0, stream>>>(h, rms_w + (size_t)l * D_MODEL, xn, D_MODEL);
    gemm_wmma_kernel<0><<<dim3((D_IN_PROJ + 63) / 64, ROWS / 64), 128, 0, stream>>>(
        xn, in_w + (size_t)l * D_MODEL * D_IN_PROJ, zx,
        D_IN_PROJ, D_MODEL, D_MODEL, D_IN_PROJ, D_IN_PROJ);
    conv_silu_dt_kernel<<<ROWS, 256, 0, stream>>>(
        zx, conv_w + (size_t)l * CONV_DIM * D_CONV, conv_b + (size_t)l * CONV_DIM,
        dt_bias + l * NHEADS, xcb, dtb);
    ssd_chunk_kernel<<<dim3(NHEADS, NCHUNK, B_SZ), 128, 0, stream>>>(
        xcb, dtb, A_log + l * NHEADS, D_param + l * NHEADS, ybuf, Sbuf, decb, acum);
    scan_kernel<<<dim3(NHEADS, B_SZ), 256, 0, stream>>>(Sbuf, decb, hsbuf);
    yoff_kernel<<<dim3(NHEADS, NCHUNK, B_SZ), 128, 0, stream>>>(xcb, hsbuf, acum, ybuf);
    gate_norm_kernel<<<ROWS, 256, 0, stream>>>(zx, gnorm_w + (size_t)l * D_INNER, ybuf);
    gemm_wmma_kernel<1><<<dim3(D_MODEL / 64, ROWS / 64), 128, 0, stream>>>(
        ybuf, out_w + (size_t)l * D_INNER * D_MODEL, h,
        D_MODEL, D_INNER, D_INNER, D_MODEL, D_MODEL);  // fused residual add
  }

  rmsnorm_kernel<<<ROWS, 256, 0, stream>>>(h, fnorm_w, xn, D_MODEL);
  head_kernel<<<ROWS, 32, 0, stream>>>(xn, head_w, head_b, out);
}
